// Attention_26920855011759
// MI455X (gfx1250) — matrix operand order repaired and checked by reading, not compiled, not run on hardware
//
#include <hip/hip_runtime.h>
#include <hip/hip_bf16.h>

// ---------------------------------------------------------------------------
// Types (trivial ext-vector types so they can live in unions)
// ---------------------------------------------------------------------------
typedef __attribute__((ext_vector_type(16))) __bf16 bf16x16;
typedef __attribute__((ext_vector_type(8)))  float  f32x8;
typedef __attribute__((ext_vector_type(4)))  unsigned int u32x4;
typedef __attribute__((ext_vector_type(4)))  int    i32x4;
typedef __attribute__((ext_vector_type(4)))  float  f32x4;

#define TILE_PITCH 40   // 32 bf16 of K data + 8 halfword pad per LDS row

// ---------------------------------------------------------------------------
// CDNA5 async global->LDS copy path (ASYNCcnt-tracked), with safe fallback
// ---------------------------------------------------------------------------
#if __has_builtin(__builtin_amdgcn_global_load_async_to_lds_b128) && \
    __has_builtin(__builtin_amdgcn_s_wait_asynccnt)
#define USE_ASYNC_COPY 1
#else
#define USE_ASYNC_COPY 0
#endif

__device__ __forceinline__ void copy_b128_g2l(const unsigned short* g,
                                              unsigned short* l) {
#if USE_ASYNC_COPY
  // Builtin expects (int4 AS(1)*, int4 AS(3)*, imm offset, imm cpol).
  // Generic LDS pointer's low 32 bits are the LDS byte offset (ISA aperture rule).
  __builtin_amdgcn_global_load_async_to_lds_b128(
      (__attribute__((address_space(1))) i32x4*)(unsigned long long)g,
      (__attribute__((address_space(3))) i32x4*)(unsigned long long)l,
      0, 0);
#else
  *(u32x4*)l = *(const u32x4*)g;
#endif
}

__device__ __forceinline__ void async_copy_fence() {
#if USE_ASYNC_COPY
  __builtin_amdgcn_s_wait_asynccnt(0);
#endif
}

__device__ __forceinline__ unsigned short f32_to_bf16(float f) {
  union { float f; unsigned int u; } v;
  v.f = f;
  unsigned int u = v.u;
  u += 0x7fffu + ((u >> 16) & 1u);   // round-to-nearest-even
  return (unsigned short)(u >> 16);
}

union FragU {
  u32x4   q[2];
  bf16x16 v;
};

// ---------------------------------------------------------------------------
// Generic bf16 WMMA GEMM:  C[m,n] = sum_k A[m,k] * B(n,k)
//   B layout: !B_KMAJOR -> B[n*ldb + k] (row = N), B_KMAJOR -> B[k*ldb + n]
//   OUT_MODE: 0 = f32 store, 1 = bf16 store, 2 = f32 tanh(acc + bias[n])
// Block tile 128(M) x 128(N) x 32(K); 256 threads = 8 waves (2 M x 4 N),
// wave tile 64x32 = 4x2 accumulators of V_WMMA_F32_16X16X32_BF16.
// Double-buffered LDS staging (async on CDNA5), one barrier per K-step.
// All dims assumed multiples of tile sizes (true for this problem).
// ---------------------------------------------------------------------------
template <int OUT_MODE, bool B_KMAJOR>
__global__ __launch_bounds__(256) void gemm_bf16_wmma(
    const unsigned short* __restrict__ A, long long strideA, int lda,
    const unsigned short* __restrict__ B, long long strideB, int ldb,
    void* __restrict__ Cv, long long strideC, int ldc,
    const float* __restrict__ bias, int K)
{
  __shared__ __align__(16) unsigned short lA[2][128 * TILE_PITCH];
  __shared__ __align__(16) unsigned short lB[2][128 * TILE_PITCH];

  const int tid   = threadIdx.x;
  const int lane  = tid & 31;
  const int wid   = tid >> 5;
  const int waveM = wid & 1;    // 2 waves along M
  const int waveN = wid >> 1;   // 4 waves along N
  const int m0    = blockIdx.y * 128;
  const int n0    = blockIdx.x * 128;
  const int batch = blockIdx.z;

  A += (long long)batch * strideA;
  B += (long long)batch * strideB;

  const int mw = waveM * 64;
  const int nw = waveN * 32;
  const int lanelo = lane & 15;
  const int lanehi = lane >> 4;

  f32x8 zacc;
#pragma unroll
  for (int i = 0; i < 8; ++i) zacc[i] = 0.f;
  f32x8 acc[4][2];
#pragma unroll
  for (int t = 0; t < 4; ++t)
#pragma unroll
    for (int s = 0; s < 2; ++s) acc[t][s] = zacc;

  // ---- staging helpers --------------------------------------------------
  auto stageA = [&](int buf, int k0) {
#pragma unroll
    for (int p = 0; p < 2; ++p) {
      int c   = tid + p * 256;      // 512 16-byte chunks (128 rows x 32 K)
      int row = c >> 2;
      int seg = (c & 3) * 8;
      copy_b128_g2l(A + (long long)(m0 + row) * lda + k0 + seg,
                    &lA[buf][row * TILE_PITCH + seg]);
    }
  };
  auto stageB = [&](int buf, int k0) {
    if (!B_KMAJOR) {
#pragma unroll
      for (int p = 0; p < 2; ++p) {
        int c   = tid + p * 256;
        int row = c >> 2;
        int seg = (c & 3) * 8;
        copy_b128_g2l(B + (long long)(n0 + row) * ldb + k0 + seg,
                      &lB[buf][row * TILE_PITCH + seg]);
      }
    } else {
      // source is [k][n]: vector-load along N, transpose-scatter into LDS
#pragma unroll
      for (int p = 0; p < 2; ++p) {
        int c    = tid + p * 256;
        int kk   = c >> 4;          // 0..31
        int nseg = (c & 15) * 8;    // 0..120
        u32x4 d = *(const u32x4*)(B + (long long)(k0 + kk) * ldb + n0 + nseg);
#pragma unroll
        for (int j = 0; j < 8; ++j) {
          unsigned int w = d[j >> 1];
          unsigned short e = (j & 1) ? (unsigned short)(w >> 16)
                                     : (unsigned short)(w & 0xffffu);
          lB[buf][(nseg + j) * TILE_PITCH + kk] = e;
        }
      }
    }
  };

  // ---- prologue: stage first K-slab ----
  stageA(0, 0);
  stageB(0, 0);
  async_copy_fence();
  __syncthreads();

  int buf = 0;
  for (int k0 = 0; k0 < K; k0 += 32) {
    const bool more = (k0 + 32 < K);
    if (more) {
      stageA(buf ^ 1, k0 + 32);
      stageB(buf ^ 1, k0 + 32);
    }
    if (k0 + 64 < K) {   // L2 prefetch two slabs ahead
      __builtin_prefetch((const void*)(A + (long long)(m0 + (tid >> 1)) * lda + k0 + 64), 0, 1);
      if (!B_KMAJOR)
        __builtin_prefetch((const void*)(B + (long long)(n0 + (tid >> 1)) * ldb + k0 + 64), 0, 1);
      else
        __builtin_prefetch((const void*)(B + (long long)(k0 + 64 + (tid & 31)) * ldb + n0), 0, 1);
    }

    // ---- fragment loads per ISA 16-bit A (16x32) / B (32x16) layouts ----
    FragU fa[4], fb[2];
#pragma unroll
    for (int t = 0; t < 4; ++t) {
      int row = mw + t * 16 + lanelo;
      int off = row * TILE_PITCH + lanehi * 8;        // K = lanehi*8 .. +7
      fa[t].q[0] = *(const u32x4*)(&lA[buf][off]);
      fa[t].q[1] = *(const u32x4*)(&lA[buf][off + 16]); // K = 16+lanehi*8 ..
    }
#pragma unroll
    for (int s = 0; s < 2; ++s) {
      int nrow = nw + s * 16 + lanelo;
      int off  = nrow * TILE_PITCH + lanehi * 8;      // K = lanehi*8 .. +7
      fb[s].q[0] = *(const u32x4*)(&lB[buf][off]);
      fb[s].q[1] = *(const u32x4*)(&lB[buf][off + 16]);
    }

#pragma unroll
    for (int t = 0; t < 4; ++t)
#pragma unroll
      for (int s = 0; s < 2; ++s)
        acc[t][s] = __builtin_amdgcn_wmma_f32_16x16x32_bf16(
            false, fa[t].v, false, fb[s].v, (short)0, acc[t][s], false, false);

    if (more) {
      async_copy_fence();   // this wave's copies into buf^1 are done
      __syncthreads();      // everyone's copies done + everyone done reading buf
      buf ^= 1;
    }
  }

  // ---- epilogue: C/D layout — VGPR r: lanes 0-15 -> M=r, lanes 16-31 -> M=8+r
#pragma unroll
  for (int t = 0; t < 4; ++t) {
#pragma unroll
    for (int s = 0; s < 2; ++s) {
#pragma unroll
      for (int r = 0; r < 8; ++r) {
        int m = m0 + mw + t * 16 + lanehi * 8 + r;
        int n = n0 + nw + s * 16 + lanelo;
        float v = acc[t][s][r];
        long long idx = (long long)batch * strideC + (long long)m * ldc + n;
        if (OUT_MODE == 0) {
          ((float*)Cv)[idx] = v;
        } else if (OUT_MODE == 1) {
          ((unsigned short*)Cv)[idx] = f32_to_bf16(v);
        } else {
          ((float*)Cv)[idx] = tanhf(v + bias[n]);
        }
      }
    }
  }
}

// ---------------------------------------------------------------------------
// Elementwise f32 -> bf16 (vectorized x4); n must be a multiple of 4
// ---------------------------------------------------------------------------
__global__ __launch_bounds__(256) void cvt_f32_bf16(
    const float* __restrict__ s, unsigned short* __restrict__ d, long long n)
{
  long long i = ((long long)blockIdx.x * blockDim.x + threadIdx.x) * 4;
  if (i + 3 < n) {
    f32x4 f = *(const f32x4*)(s + i);
#pragma unroll
    for (int j = 0; j < 4; ++j) d[i + j] = f32_to_bf16(f[j]);
  }
}

// output f32 [16384,1024] -> bf16 into cat[:, 1024:2048] (ld 2048)
__global__ __launch_bounds__(256) void pack_output_kernel(
    const float* __restrict__ src, unsigned short* __restrict__ cat)
{
  long long i = (long long)blockIdx.x * blockDim.x + threadIdx.x; // < 16777216
  long long row = i >> 10;
  int col = (int)(i & 1023);
  cat[row * 2048 + 1024 + col] = f32_to_bf16(src[i]);
}

// ---------------------------------------------------------------------------
// Mask + softmax over rows of length 2048 (in place on f32 attn, bf16 copy)
// ---------------------------------------------------------------------------
__global__ __launch_bounds__(256) void softmax_mask_kernel(
    float* __restrict__ attn, unsigned short* __restrict__ attn_bf,
    const int* __restrict__ masks)
{
  const int row = blockIdx.x;                 // b*Y + y
  const long long base = (long long)row * 2048;
  const int tid = threadIdx.x;
  __shared__ float red[8];

  float v[8];
  float mx = -3.4e38f;
#pragma unroll
  for (int i = 0; i < 8; ++i) {
    long long idx = base + (long long)tid * 8 + i;
    float x = attn[idx];
    if (masks[idx] != 0) x = -1e-6f;          // faithful to reference quirk
    v[i] = x;
    mx = fmaxf(mx, x);
  }
#pragma unroll
  for (int off = 16; off > 0; off >>= 1)
    mx = fmaxf(mx, __shfl_xor(mx, off, 32));
  if ((tid & 31) == 0) red[tid >> 5] = mx;
  __syncthreads();
  float rmax = red[0];
#pragma unroll
  for (int w = 1; w < 8; ++w) rmax = fmaxf(rmax, red[w]);
  __syncthreads();

  float sum = 0.f;
#pragma unroll
  for (int i = 0; i < 8; ++i) { v[i] = __expf(v[i] - rmax); sum += v[i]; }
#pragma unroll
  for (int off = 16; off > 0; off >>= 1)
    sum += __shfl_xor(sum, off, 32);
  if ((tid & 31) == 0) red[tid >> 5] = sum;
  __syncthreads();
  float tot = 0.f;
#pragma unroll
  for (int w = 0; w < 8; ++w) tot += red[w];
  float inv = 1.0f / tot;

#pragma unroll
  for (int i = 0; i < 8; ++i) {
    long long idx = base + (long long)tid * 8 + i;
    float o = v[i] * inv;
    attn[idx]    = o;
    attn_bf[idx] = f32_to_bf16(o);
  }
}

// ---------------------------------------------------------------------------
// Host launch. Sizes: B=8, Y=X=2048, E2=D=1024.
// d_out = out f32 [8*2048*1024] ++ atten f32 [8*2048*2048]
// ---------------------------------------------------------------------------
extern "C" void kernel_launch(void* const* d_in, const int* in_sizes, int n_in,
                              void* d_out, int out_size, void* d_ws, size_t ws_size,
                              hipStream_t stream) {
  (void)in_sizes; (void)n_in; (void)out_size; (void)ws_size;

  const float* outp   = (const float*)d_in[0];  // [8,2048,1024]
  const float* contex = (const float*)d_in[1];  // [8,2048,1024]
  const int*   masks  = (const int*)  d_in[2];  // [8,2048,2048]
  const float* W_in   = (const float*)d_in[3];  // [1024,1024]
  const float* W_out  = (const float*)d_in[4];  // [1024,2048]
  const float* b_out  = (const float*)d_in[5];  // [1024]

  float* out_final = (float*)d_out;             // 16777216 f32
  float* attn      = out_final + 16777216LL;    // 33554432 f32

  // workspace layout (bf16 as u16), ~198 MB total
  unsigned short* ctx_bf   = (unsigned short*)d_ws;        // 16777216
  unsigned short* ctxin_bf = ctx_bf   + 16777216LL;        // 16777216
  unsigned short* cat_bf   = ctxin_bf + 16777216LL;        // 33554432 ([.,0:1024]=ctx, [.,1024:2048]=output)
  unsigned short* attn_bf  = cat_bf   + 33554432LL;        // 33554432
  unsigned short* win_bf   = attn_bf  + 33554432LL;        // 1048576
  unsigned short* wout_bf  = win_bf   + 1048576LL;         // 2097152

  // precision conversion passes
  cvt_f32_bf16<<<16777216 / 1024, 256, 0, stream>>>(contex, ctx_bf, 16777216LL);
  cvt_f32_bf16<<<1048576  / 1024, 256, 0, stream>>>(W_in,   win_bf, 1048576LL);
  cvt_f32_bf16<<<2097152  / 1024, 256, 0, stream>>>(W_out,  wout_bf, 2097152LL);
  pack_output_kernel<<<65536, 256, 0, stream>>>(outp, cat_bf);

  // GEMM1: context_in[bx,d] = sum_e context[bx,e] * W_in[d,e]   (M=16384,N=1024,K=1024)
  gemm_bf16_wmma<1, false><<<dim3(8, 128, 1), 256, 0, stream>>>(
      ctx_bf, 0, 1024, win_bf, 0, 1024, (void*)ctxin_bf, 0, 1024, nullptr, 1024);

  // GEMM2: attn_raw[b,y,x] = sum_d output[b,y,d] * context_in[b,x,d]  (per-batch)
  gemm_bf16_wmma<0, false><<<dim3(16, 16, 8), 256, 0, stream>>>(
      cat_bf + 1024, 2048LL * 2048, 2048,
      ctxin_bf,      2048LL * 1024, 1024,
      (void*)attn,   2048LL * 2048, 2048, nullptr, 1024);

  // mask + softmax (writes f32 output attn + bf16 copy)
  softmax_mask_kernel<<<16384, 256, 0, stream>>>(attn, attn_bf, masks);

  // GEMM3: ctx[b,y,e] = sum_x attn[b,y,x] * context[b,x,e]   (B is K-major)
  gemm_bf16_wmma<1, true><<<dim3(8, 16, 8), 256, 0, stream>>>(
      attn_bf,       2048LL * 2048, 2048,
      ctx_bf,        2048LL * 1024, 1024,
      (void*)cat_bf, 2048LL * 2048, 2048, nullptr, 2048);

  // GEMM4: out = tanh(cat @ W_out^T + b_out)   (M=16384,N=1024,K=2048)
  gemm_bf16_wmma<2, false><<<dim3(8, 128, 1), 256, 0, stream>>>(
      cat_bf, 0, 2048, wout_bf, 0, 2048,
      (void*)out_final, 0, 1024, b_out, 2048);
}
